// Seq2Seq_9285719294072
// MI455X (gfx1250) — compile-verified, matrix operand order checked by reading, not run on hardware
//
#include <hip/hip_runtime.h>
#include <hip/hip_bf16.h>
#include <stdint.h>

// Problem dims (from reference)
#define B_    64
#define SSRC_ 100
#define T_    80
#define ED_   512
#define ZD_   512
#define EH_   512
#define ZH_   1024
#define A_    512
#define EV_   30000
#define ZV_   4004
#define DIN_  1536   // 2*EH + ZD

typedef __bf16 bf16_t;
typedef __attribute__((ext_vector_type(16))) __bf16        v16bf;
typedef __attribute__((ext_vector_type(8)))  float         v8f;
typedef __attribute__((ext_vector_type(4)))  unsigned int  u32x4;

__device__ __forceinline__ unsigned short f32_bf16(float f) {
  unsigned int u = __builtin_bit_cast(unsigned int, f);
  u += 0x7FFFu + ((u >> 16) & 1u);           // round-to-nearest-even
  return (unsigned short)(u >> 16);
}

__device__ __forceinline__ float sigmoidf_(float x) {
  return 1.f / (1.f + expf(-x));
}

union Frag { v16bf v; u32x4 q[2]; };

// ---------------- elementwise / conversion kernels ----------------

__global__ void cvt_f32_to_bf16(const float* __restrict__ s, unsigned short* __restrict__ d, int n) {
  int i = blockIdx.x * blockDim.x + threadIdx.x;
  if (i < n) d[i] = f32_bf16(s[i]);
}

__global__ void gather_emb_bf16(const int* __restrict__ idx, const float* __restrict__ emb,
                                unsigned short* __restrict__ dst, int rows, int width) {
  int i = blockIdx.x * blockDim.x + threadIdx.x;
  if (i >= rows * width) return;
  int r = i / width, c = i - r * width;
  dst[i] = f32_bf16(emb[(size_t)idx[r] * width + c]);
}

__global__ void zero_f32(float* p, int n) { int i = blockIdx.x*blockDim.x+threadIdx.x; if (i < n) p[i] = 0.f; }
__global__ void zero_u16(unsigned short* p, int n) { int i = blockIdx.x*blockDim.x+threadIdx.x; if (i < n) p[i] = 0; }

// ---------------- WMMA GEMM ----------------
// C[M,N] = A1[M,K1] @ W1[N,K1]^T (+ A2[M,K2] @ W2[N,K2]^T) + bias1[N] + bias2[N]
// A*/W* bf16 (as u16), C fp32.
// Each wave computes an (MB*16) x 16 tile: the B (weight) fragment is loaded once
// per k-chunk and reused across MB accumulators -> weight/L2 traffic divided by MB.
// k-loop is explicitly double-buffered with two STATIC fragment register sets
// (k unrolled by 64; all K here are multiples of 64) so next-chunk loads are in
// flight while current-chunk WMMAs issue. __launch_bounds__(128) lifts the
// default 1024-thread VGPR cap so MB=4 fits in registers without scratch spills.
// Fragment layout (ISA 7.12.2, 16-bit A 16x32): lane = row (M for A, N for W^T),
// half-wave selects K groups {0..7,16..23} / {8..15,24..31} -> two b128 loads.
template <int MB>
__device__ __forceinline__ void load_chunk(const unsigned short* __restrict__ wp,
                                           const unsigned short* __restrict__ ap0,
                                           size_t amstride,
                                           int k0, int half, Frag& fb, Frag* fa)
{
  const u32x4* pw = (const u32x4*)(wp + k0);
  __builtin_prefetch((const void*)(wp + k0 + 512), 0, 1);
  fb.q[0] = pw[half];  fb.q[1] = pw[half + 2];
  #pragma unroll
  for (int mi = 0; mi < MB; ++mi) {
    const u32x4* pa = (const u32x4*)(ap0 + (size_t)mi * amstride + k0);
    fa[mi].q[0] = pa[half];  fa[mi].q[1] = pa[half + 2];
  }
}

template <int MB>
__device__ __forceinline__ void gemm_accum(const unsigned short* __restrict__ A, int lda,
                                           const unsigned short* __restrict__ W, int ldw, int K,
                                           int arow0, int wrow, int half,
                                           v8f* acc)
{
  const unsigned short* wp  = W + (size_t)wrow * ldw;
  const unsigned short* ap0 = A + (size_t)arow0 * lda;
  const size_t amstride = (size_t)16 * lda;   // rows arow0 + mi*16 (M % (16*MB) == 0 here)

  Frag fb0, fb1;
  Frag fa0[MB], fa1[MB];
  load_chunk<MB>(wp, ap0, amstride, 0, half, fb0, fa0);
  for (int k0 = 0; k0 < K; k0 += 64) {        // K % 64 == 0 (512/1024/1536)
    load_chunk<MB>(wp, ap0, amstride, k0 + 32, half, fb1, fa1);
    #pragma unroll
    for (int mi = 0; mi < MB; ++mi)
      acc[mi] = __builtin_amdgcn_wmma_f32_16x16x32_bf16(false, fa0[mi].v, false, fb0.v,
                                                        (short)0, acc[mi], false, false);
    if (k0 + 64 < K)
      load_chunk<MB>(wp, ap0, amstride, k0 + 64, half, fb0, fa0);
    #pragma unroll
    for (int mi = 0; mi < MB; ++mi)
      acc[mi] = __builtin_amdgcn_wmma_f32_16x16x32_bf16(false, fa1[mi].v, false, fb1.v,
                                                        (short)0, acc[mi], false, false);
  }
}

template <int MB>
__global__ __launch_bounds__(128)
void gemm_bf16_wmma(const unsigned short* __restrict__ A1, int lda1,
                    const unsigned short* __restrict__ W1, int ldw1, int K1,
                    const unsigned short* __restrict__ A2, int lda2,
                    const unsigned short* __restrict__ W2, int ldw2, int K2,
                    float* __restrict__ C, int ldc, int M, int N,
                    const float* __restrict__ bias1,
                    const float* __restrict__ bias2)
{
  const int lane = threadIdx.x & 31;
  const int wave = threadIdx.x >> 5;
  const int n0  = (blockIdx.x * 4 + wave) * 16;
  const int mb0 = blockIdx.y * (16 * MB);
  if (n0 >= N || mb0 >= M) return;         // wave-uniform exit: EXEC stays all-1 for WMMA
  const int half = lane >> 4;
  const int r    = lane & 15;
  const int wrow = (n0 + r < N) ? (n0 + r) : (N - 1);   // clamp keeps EXEC full
  const int arow0 = mb0 + r;               // M is a multiple of 16*MB for all calls

  v8f acc[MB];
  const v8f vzero = {0.f,0.f,0.f,0.f,0.f,0.f,0.f,0.f};
  #pragma unroll
  for (int mi = 0; mi < MB; ++mi) acc[mi] = vzero;

  gemm_accum<MB>(A1, lda1, W1, ldw1, K1, arow0, wrow, half, acc);
  if (A2 != nullptr)
    gemm_accum<MB>(A2, lda2, W2, ldw2, K2, arow0, wrow, half, acc);

  // C/D layout: acc[mi][i] -> row mb0 + mi*16 + i + 8*half, col n0 + (lane&15)
  const int col = n0 + r;
  if (col < N) {
    float bsum = 0.f;
    if (bias1) bsum += bias1[col];
    if (bias2) bsum += bias2[col];
    #pragma unroll
    for (int mi = 0; mi < MB; ++mi) {
      #pragma unroll
      for (int i = 0; i < 8; ++i) {
        int row = mb0 + mi * 16 + i + 8 * half;
        if (row < M) C[(size_t)row * ldc + col] = acc[mi][i] + bsum;
      }
    }
  }
}

// ---------------- LSTM gate fusion ----------------
// g layout: [B][4H] with gate order i,f,g,o. Applies masking (t < lens[b]) when lens!=null.
// Writes fp32 state + bf16 state, and optionally the per-step output slice (fp32 + bf16).
__global__ void lstm_gates(const float* __restrict__ g, int H,
                           float* __restrict__ h, float* __restrict__ c,
                           unsigned short* __restrict__ h_bf,
                           float* __restrict__ out_f, unsigned short* __restrict__ out_bf,
                           long long out_bstride,
                           const int* __restrict__ lens, int t)
{
  int i = blockIdx.x * blockDim.x + threadIdx.x;
  if (i >= B_ * H) return;
  int b = i / H, j = i - b * H;
  const float* gb = g + (size_t)b * 4 * H;
  float gi = gb[j], gf = gb[H + j], gg = gb[2*H + j], go = gb[3*H + j];
  float cn = sigmoidf_(gf) * c[i] + sigmoidf_(gi) * tanhf(gg);
  float hn = sigmoidf_(go) * tanhf(cn);
  float m  = (lens == nullptr) ? 1.f : ((t < lens[b]) ? 1.f : 0.f);
  float hN = m * hn + (1.f - m) * h[i];
  float cN = m * cn + (1.f - m) * c[i];
  h[i] = hN;  c[i] = cN;  h_bf[i] = f32_bf16(hN);
  if (out_f) {
    float o = m * hn;
    out_f [(size_t)b * out_bstride + j] = o;
    out_bf[(size_t)b * out_bstride + j] = f32_bf16(o);
  }
}

// ---------------- attention ----------------
__global__ void attn_score(const float* __restrict__ Uh, const float* __restrict__ wsb,
                           const float* __restrict__ Wv, const float* __restrict__ bv,
                           float* __restrict__ score)
{
  int s = blockIdx.x, b = blockIdx.y;
  const float* u = Uh  + ((size_t)b * SSRC_ + s) * A_;
  const float* w = wsb + (size_t)b * A_;
  float p = 0.f;
  for (int a = threadIdx.x; a < A_; a += blockDim.x)
    p += Wv[a] * tanhf(u[a] + w[a]);
  __shared__ float red[128];
  red[threadIdx.x] = p; __syncthreads();
  for (int off = 64; off > 0; off >>= 1) {
    if (threadIdx.x < off) red[threadIdx.x] += red[threadIdx.x + off];
    __syncthreads();
  }
  if (threadIdx.x == 0) score[(size_t)b * SSRC_ + s] = red[0] + bv[0];
}

// softmax over SSRC, context att = alpha@enc, pack [att | zh_emb[gtruths[b,t]]] as bf16 input row
__global__ void attn_combine(const float* __restrict__ score, const float* __restrict__ enc,
                             const float* __restrict__ zh_emb, const int* __restrict__ gtruths,
                             int t, unsigned short* __restrict__ inp_bf)
{
  int b = blockIdx.x, tid = threadIdx.x;
  __shared__ float alpha[SSRC_];
  __shared__ float red[256];
  float v = (tid < SSRC_) ? score[(size_t)b * SSRC_ + tid] : -3.4e38f;
  red[tid] = v; __syncthreads();
  for (int off = 128; off > 0; off >>= 1) { if (tid < off) red[tid] = fmaxf(red[tid], red[tid+off]); __syncthreads(); }
  float mx = red[0]; __syncthreads();
  float e = (tid < SSRC_) ? expf(v - mx) : 0.f;
  red[tid] = e; __syncthreads();
  for (int off = 128; off > 0; off >>= 1) { if (tid < off) red[tid] += red[tid+off]; __syncthreads(); }
  float inv = 1.f / red[0];
  if (tid < SSRC_) alpha[tid] = e * inv;
  __syncthreads();
  for (int h = tid; h < 2*EH_; h += blockDim.x) {
    const float* eb = enc + (size_t)b * SSRC_ * (2*EH_) + h;
    float acc = 0.f;
    for (int s = 0; s < SSRC_; ++s) acc += alpha[s] * eb[(size_t)s * (2*EH_)];
    inp_bf[(size_t)b * DIN_ + h] = f32_bf16(acc);
  }
  int gi = gtruths[b * T_ + t];
  for (int j = tid; j < ZD_; j += blockDim.x)
    inp_bf[(size_t)b * DIN_ + 2*EH_ + j] = f32_bf16(zh_emb[(size_t)gi * ZD_ + j]);
}

__global__ void argmax_pred(const float* __restrict__ logits, float* __restrict__ pred, int t)
{
  int b = blockIdx.x, tid = threadIdx.x;
  const float* l = logits + (size_t)b * ZV_;
  float best = -3.4e38f; int bi = 0;
  for (int v = tid; v < ZV_; v += blockDim.x) {
    float x = l[v];
    if (x > best) { best = x; bi = v; }
  }
  __shared__ float rv[256]; __shared__ int ri[256];
  rv[tid] = best; ri[tid] = bi; __syncthreads();
  for (int off = 128; off > 0; off >>= 1) {
    if (tid < off) {
      if (rv[tid+off] > rv[tid] || (rv[tid+off] == rv[tid] && ri[tid+off] < ri[tid])) {
        rv[tid] = rv[tid+off]; ri[tid] = ri[tid+off];
      }
    }
    __syncthreads();
  }
  if (tid == 0) pred[b * T_ + t] = (float)ri[0];
}

// ---------------- host orchestration ----------------

extern "C" void kernel_launch(void* const* d_in, const int* in_sizes, int n_in,
                              void* d_out, int out_size, void* d_ws, size_t ws_size,
                              hipStream_t stream)
{
  const int*   inputs  = (const int*)  d_in[0];
  const int*   gtruths = (const int*)  d_in[1];
  const int*   lens    = (const int*)  d_in[2];
  const float* en_emb  = (const float*)d_in[3];
  const float* zh_emb  = (const float*)d_in[4];
  const float* Wih_f   = (const float*)d_in[5];
  const float* Whh_f   = (const float*)d_in[6];
  const float* bih_f   = (const float*)d_in[7];
  const float* bhh_f   = (const float*)d_in[8];
  const float* Wih_b   = (const float*)d_in[9];
  const float* Whh_b   = (const float*)d_in[10];
  const float* bih_b   = (const float*)d_in[11];
  const float* bhh_b   = (const float*)d_in[12];
  const float* Wih_d   = (const float*)d_in[13];
  const float* Whh_d   = (const float*)d_in[14];
  const float* bih_d   = (const float*)d_in[15];
  const float* bhh_d   = (const float*)d_in[16];
  const float* Wws     = (const float*)d_in[17];
  const float* bws     = (const float*)d_in[18];
  const float* Wuh     = (const float*)d_in[19];
  const float* buh     = (const float*)d_in[20];
  const float* Wv      = (const float*)d_in[21];
  const float* bv      = (const float*)d_in[22];
  const float* Wfc     = (const float*)d_in[23];
  const float* bfc     = (const float*)d_in[24];
  float* out = (float*)d_out;

  // workspace carve-out (deterministic; ~101 MB total)
  char* base = (char*)d_ws; size_t pos = 0;
  auto alloc = [&](size_t bytes) -> void* {
    void* p = base + pos; pos = (pos + bytes + 255) & ~(size_t)255; return p;
  };
  unsigned short* bWih_f = (unsigned short*)alloc((size_t)4*EH_*ED_*2);
  unsigned short* bWhh_f = (unsigned short*)alloc((size_t)4*EH_*EH_*2);
  unsigned short* bWih_b = (unsigned short*)alloc((size_t)4*EH_*ED_*2);
  unsigned short* bWhh_b = (unsigned short*)alloc((size_t)4*EH_*EH_*2);
  unsigned short* bWih_d = (unsigned short*)alloc((size_t)4*ZH_*DIN_*2);
  unsigned short* bWhh_d = (unsigned short*)alloc((size_t)4*ZH_*ZH_*2);
  unsigned short* bWfc   = (unsigned short*)alloc((size_t)ZV_*ZH_*2);
  unsigned short* bWws   = (unsigned short*)alloc((size_t)A_*ZH_*2);
  unsigned short* bWuh   = (unsigned short*)alloc((size_t)A_*2*EH_*2);
  unsigned short* Xbf    = (unsigned short*)alloc((size_t)B_*SSRC_*ED_*2);
  float*          enc_f  = (float*)         alloc((size_t)B_*SSRC_*2*EH_*4);
  unsigned short* enc_bf = (unsigned short*)alloc((size_t)B_*SSRC_*2*EH_*2);
  float*          Uh     = (float*)         alloc((size_t)B_*SSRC_*A_*4);
  float*          gbuf   = (float*)         alloc((size_t)B_*4*ZH_*4);
  float*          stf    = (float*)         alloc((size_t)(4*B_*EH_ + 2*B_*ZH_)*4); // hf,cf,hb,cb,hd,cd
  unsigned short* stb    = (unsigned short*)alloc((size_t)(2*B_*EH_ + B_*ZH_)*2);   // hf_bf,hb_bf,hd_bf
  float*          wsb    = (float*)         alloc((size_t)B_*A_*4);
  float*          scoreb = (float*)         alloc((size_t)B_*SSRC_*4);
  unsigned short* inp_bf = (unsigned short*)alloc((size_t)B_*DIN_*2);

  float *hf = stf,              *cf = hf + B_*EH_,
        *hb = cf + B_*EH_,      *cb = hb + B_*EH_,
        *hd = cb + B_*EH_,      *cd = hd + B_*ZH_;
  unsigned short *hf_bf = stb, *hb_bf = hf_bf + B_*EH_, *hd_bf = hb_bf + B_*EH_;

  auto cvt = [&](const float* s, unsigned short* d, int n) {
    cvt_f32_to_bf16<<<(n + 255)/256, 256, 0, stream>>>(s, d, n);
  };
  auto gemm = [&](const unsigned short* A1, int lda1, const unsigned short* W1, int ldw1, int K1,
                  const unsigned short* A2, int lda2, const unsigned short* W2, int ldw2, int K2,
                  float* C, int ldc, int M, int N, const float* b1, const float* b2) {
    if (M >= 1024) {        // big GEMM (Uh): plenty of M-parallelism -> full 4x weight reuse
      dim3 grid((N + 63) / 64, (M + 63) / 64);
      gemm_bf16_wmma<4><<<grid, 128, 0, stream>>>(A1, lda1, W1, ldw1, K1,
                                                  A2, lda2, W2, ldw2, K2,
                                                  C, ldc, M, N, b1, b2);
    } else {                // per-step GEMMs (M=64): balance reuse vs wave count
      dim3 grid((N + 63) / 64, (M + 31) / 32);
      gemm_bf16_wmma<2><<<grid, 128, 0, stream>>>(A1, lda1, W1, ldw1, K1,
                                                  A2, lda2, W2, ldw2, K2,
                                                  C, ldc, M, N, b1, b2);
    }
  };

  // 1) weights -> bf16 (stays hot in 192MB L2 for the whole sequential part)
  cvt(Wih_f, bWih_f, 4*EH_*ED_);   cvt(Whh_f, bWhh_f, 4*EH_*EH_);
  cvt(Wih_b, bWih_b, 4*EH_*ED_);   cvt(Whh_b, bWhh_b, 4*EH_*EH_);
  cvt(Wih_d, bWih_d, 4*ZH_*DIN_);  cvt(Whh_d, bWhh_d, 4*ZH_*ZH_);
  cvt(Wfc,   bWfc,   ZV_*ZH_);     cvt(Wws,   bWws,   A_*ZH_);
  cvt(Wuh,   bWuh,   A_*2*EH_);

  // 2) embed encoder inputs, zero states
  gather_emb_bf16<<<((B_*SSRC_*ED_) + 255)/256, 256, 0, stream>>>(inputs, en_emb, Xbf, B_*SSRC_, ED_);
  { int nf = 4*B_*EH_ + 2*B_*ZH_; zero_f32<<<(nf+255)/256, 256, 0, stream>>>(stf, nf); }
  { int nu = 2*B_*EH_ + B_*ZH_;   zero_u16<<<(nu+255)/256, 256, 0, stream>>>(stb, nu); }

  const long long encStride = (long long)SSRC_ * 2 * EH_;

  // 3) encoder: forward LSTM
  for (int t = 0; t < SSRC_; ++t) {
    gemm(Xbf + (size_t)t*ED_, SSRC_*ED_, bWih_f, ED_, ED_,
         hf_bf, EH_, bWhh_f, EH_, EH_,
         gbuf, 4*EH_, B_, 4*EH_, bih_f, bhh_f);
    lstm_gates<<<(B_*EH_+255)/256, 256, 0, stream>>>(gbuf, EH_, hf, cf, hf_bf,
        enc_f + (size_t)t*2*EH_, enc_bf + (size_t)t*2*EH_, encStride, lens, t);
  }
  // 4) encoder: backward LSTM (recurrence high->low t, same masking semantics)
  for (int t = SSRC_ - 1; t >= 0; --t) {
    gemm(Xbf + (size_t)t*ED_, SSRC_*ED_, bWih_b, ED_, ED_,
         hb_bf, EH_, bWhh_b, EH_, EH_,
         gbuf, 4*EH_, B_, 4*EH_, bih_b, bhh_b);
    lstm_gates<<<(B_*EH_+255)/256, 256, 0, stream>>>(gbuf, EH_, hb, cb, hb_bf,
        enc_f + (size_t)t*2*EH_ + EH_, enc_bf + (size_t)t*2*EH_ + EH_, encStride, lens, t);
  }

  // 5) Uh = enc @ Wuh^T + buh  (one big 6400x512x1024 WMMA GEMM)
  gemm(enc_bf, 2*EH_, bWuh, 2*EH_, 2*EH_, nullptr, 0, nullptr, 0, 0,
       Uh, A_, B_*SSRC_, A_, buh, nullptr);

  // 6) decoder
  for (int t = 0; t < T_; ++t) {
    gemm(hd_bf, ZH_, bWws, ZH_, ZH_, nullptr, 0, nullptr, 0, 0,
         wsb, A_, B_, A_, bws, nullptr);
    attn_score<<<dim3(SSRC_, B_), 128, 0, stream>>>(Uh, wsb, Wv, bv, scoreb);
    attn_combine<<<B_, 256, 0, stream>>>(scoreb, enc_f, zh_emb, gtruths, t, inp_bf);
    gemm(inp_bf, DIN_, bWih_d, DIN_, DIN_,
         hd_bf, ZH_, bWhh_d, ZH_, ZH_,
         gbuf, 4*ZH_, B_, 4*ZH_, bih_d, bhh_d);
    lstm_gates<<<(B_*ZH_+255)/256, 256, 0, stream>>>(gbuf, ZH_, hd, cd, hd_bf,
        nullptr, nullptr, 0, nullptr, 0);
    float* lt = out + (size_t)t * B_ * ZV_;
    gemm(hd_bf, ZH_, bWfc, ZH_, ZH_, nullptr, 0, nullptr, 0, 0,
         lt, ZV_, B_, ZV_, bfc, nullptr);
    argmax_pred<<<B_, 256, 0, stream>>>(lt, out + (size_t)T_ * B_ * ZV_, t);
  }
}
